// InversePredictionModel_14886356648152
// MI455X (gfx1250) — matrix operand order repaired and checked by reading, not compiled, not run on hardware
//
#include <hip/hip_runtime.h>
#include <hip/hip_bf16.h>
#include <stdint.h>

// ---------------- problem constants ----------------
#define N_SPOTS 50000
#define N_GENES 2000
#define HID     64
#define BN_EPS  1e-5f
#define MTILES  (N_SPOTS / 16)        // 3125 (exact)
#define KT_X    ((N_GENES + 31) / 32) // 63 k-tiles for K=2000 (zero padded)
#define KT_H    (HID / 32)            // 2
#define NT_H    (HID / 16)            // 4
#define NT_G    (N_GENES / 16)        // 125

typedef __attribute__((ext_vector_type(16))) __bf16 bf16x16;
typedef __attribute__((ext_vector_type(2)))  __bf16 bf16x2;
typedef __attribute__((ext_vector_type(8)))  float  floatx8;

union FragU { unsigned u[8]; uint4 q[2]; bf16x16 v; };

// pack two f32 -> packed bf16 dword.
// Preferred: cvt_pk_bf16_f32 builtin; else 2x v_add + 1x v_perm_b32 (round-to-nearest, ties away).
__device__ __forceinline__ unsigned packbf(float a, float b) {
#if __has_builtin(__builtin_amdgcn_cvt_pk_bf16_f32)
  bf16x2 p = __builtin_amdgcn_cvt_pk_bf16_f32(a, b);
  return __builtin_bit_cast(unsigned, p);
#elif __has_builtin(__builtin_amdgcn_perm)
  unsigned ua = __builtin_bit_cast(unsigned, a) + 0x8000u;
  unsigned ub = __builtin_bit_cast(unsigned, b) + 0x8000u;
  // result = { ub[31:16], ua[31:16] } : bytes ua.2, ua.3, ub.2, ub.3
  return __builtin_amdgcn_perm(ub, ua, 0x07060302u);
#else
  unsigned ua = __builtin_bit_cast(unsigned, a) + 0x8000u;
  unsigned ub = __builtin_bit_cast(unsigned, b) + 0x8000u;
  return (ua >> 16) | (ub & 0xffff0000u);
#endif
}
__device__ __forceinline__ unsigned short f2bf(float f) {
  return (unsigned short)((__builtin_bit_cast(unsigned, f) + 0x8000u) >> 16);
}
__device__ __forceinline__ floatx8 zero8() {
  floatx8 z = {0.f, 0.f, 0.f, 0.f, 0.f, 0.f, 0.f, 0.f};
  return z;
}
__device__ __forceinline__ floatx8 wmma_bf16(const FragU& a, const FragU& b, floatx8 c) {
  // D = A(16x32 bf16) * B(32x16 bf16) + C(16x16 f32)
  return __builtin_amdgcn_wmma_f32_16x16x32_bf16(false, a.v, false, b.v,
                                                 (short)0, c, false, false);
}
// WGP-scope prefetch: SCOPE 0 pulls the line into ALL cache levels (L0 + GL2),
// unlike __builtin_prefetch which lowers to SCOPE_SE (GL2 only).
__device__ __forceinline__ void prefetch_wgp(const void* p) {
  asm volatile("global_prefetch_b8 %0, off" :: "v"(p));
}

// ---------------- utility kernels ----------------
__global__ void k_zero_f32(float* __restrict__ p, int n) {
  int i = blockIdx.x * blockDim.x + threadIdx.x;
  int stride = gridDim.x * blockDim.x;
  for (; i < n; i += stride) p[i] = 0.f;
}

// Swizzle a row-major f32 weight [Kdim, Ndim] into bf16 WMMA B fragments.
// Fragment layout: dword index = ((nt*KT + t)*32 + lane)*8 + j
//   lane 0-15  -> N = nt*16 + lane,      K = t*32 + {0-7, 16-23} (pairs per j)
//   lane 16-31 -> N = nt*16 + lane-16,   K = t*32 + {8-15, 24-31} (pairs per j)
__global__ void k_build_frag(const float* __restrict__ W, int Kdim, int Ndim,
                             int KT, int NT, unsigned* __restrict__ frag) {
  int total = KT * NT * 256;
  int i = blockIdx.x * blockDim.x + threadIdx.x;
  int stride = gridDim.x * blockDim.x;
  for (; i < total; i += stride) {
    int j    = i & 7;
    int lane = (i >> 3) & 31;
    int rem  = i >> 8;
    int t    = rem % KT;
    int nt   = rem / KT;
    int k0 = t * 32 + ((lane & 16) ? 8 : 0) + ((j < 4) ? 2 * j : 16 + 2 * (j - 4));
    int n  = nt * 16 + (lane & 15);
    float v0 = (k0     < Kdim && n < Ndim) ? W[(size_t)k0       * Ndim + n] : 0.f;
    float v1 = (k0 + 1 < Kdim && n < Ndim) ? W[(size_t)(k0 + 1) * Ndim + n] : 0.f;
    frag[i] = packbf(v0, v1);
  }
}

// Load bf16 A-fragment for one 16x32 k-tile from a row-major bf16 [rows, 64] buffer.
// A layout (16-bit, 16x32): lanes 0-15 hold M=lane, K 0-7 & 16-23;
//                           lanes 16-31 hold M=lane-16, K 8-15 & 24-31.
__device__ __forceinline__ void load_afrag_bf16(const unsigned short* __restrict__ rowp,
                                                int ktile, int khalf, FragU& a) {
  int c0 = ktile * 32 + khalf;               // khalf = (lane&16)?8:0
  a.q[0] = *(const uint4*)(rowp + c0);       // K = c0 .. c0+7   (16B, aligned)
  a.q[1] = *(const uint4*)(rowp + c0 + 16);  // K = c0+16 .. c0+23
}

__device__ __forceinline__ void load_bfrag(const unsigned* __restrict__ wf,
                                           int nt, int t, int KT, int lane, FragU& b) {
  const uint4* p = (const uint4*)(wf + ((((size_t)nt * KT + t) * 32) + lane) * 8);
  b.q[0] = p[0];
  b.q[1] = p[1];
}

// ---------------- z = x @ W1 + b1  (the 400MB-read GEMM) ----------------
// launch_bounds min-occupancy 1: free register budget so all 4 B fragments
// can stay live -> one load clause + one wait per k-tile instead of four.
__global__ __launch_bounds__(256, 1) void k_gemm1(const float* __restrict__ x,
                                                  const unsigned* __restrict__ wf,
                                                  const float* __restrict__ b1,
                                                  float* __restrict__ z) {
  int wave = threadIdx.x >> 5, lane = threadIdx.x & 31;
  int mtile = blockIdx.x * 8 + wave;
  if (mtile >= MTILES) return;
  int arow = mtile * 16 + (lane & 15);
  const float* xr = x + (size_t)arow * N_GENES;
  int khalf = (lane & 16) ? 8 : 0;

  floatx8 acc[NT_H];
#pragma unroll
  for (int i = 0; i < NT_H; ++i) acc[i] = zero8();

  for (int t = 0; t < KT_X; ++t) {
    int c0 = t * 32 + khalf;

    // one clause of 8 b128 loads: all 4 B fragments for this k-tile
    FragU bfr[NT_H];
#pragma unroll
    for (int nt = 0; nt < NT_H; ++nt) load_bfrag(wf, nt, t, KT_X, lane, bfr[nt]);

    // WGP-scope prefetch of NEXT k-tile's B fragments and x chunk: next
    // iteration's loads hit L0 instead of paying L2 latency per n-tile.
    if (t + 1 < KT_X) {
#pragma unroll
      for (int nt = 0; nt < NT_H; ++nt)
        prefetch_wgp(wf + ((((size_t)nt * KT_X + (t + 1)) * 32) + lane) * 8);
      prefetch_wgp(xr + c0 + 32);
    }

    // x chunk loads + bf16 pack (VALU overlaps B-load latency)
    FragU a;
#pragma unroll
    for (int h = 0; h < 2; ++h) {
      int c = c0 + h * 16;
      if (c + 8 <= N_GENES) {
        float4 p0 = ((const float4*)(xr + c))[0];
        float4 p1 = ((const float4*)(xr + c))[1];
        a.u[h * 4 + 0] = packbf(p0.x, p0.y);
        a.u[h * 4 + 1] = packbf(p0.z, p0.w);
        a.u[h * 4 + 2] = packbf(p1.x, p1.y);
        a.u[h * 4 + 3] = packbf(p1.z, p1.w);
      } else {
#pragma unroll
        for (int j = 0; j < 4; ++j) {
          int k = c + 2 * j;
          float v0 = (k     < N_GENES) ? xr[k]     : 0.f;
          float v1 = (k + 1 < N_GENES) ? xr[k + 1] : 0.f;
          a.u[h * 4 + j] = packbf(v0, v1);
        }
      }
    }

    // back-to-back WMMAs
#pragma unroll
    for (int nt = 0; nt < NT_H; ++nt)
      acc[nt] = wmma_bf16(a, bfr[nt], acc[nt]);
  }

  // C layout: VGPR r -> row = r + 8*(lane>=16), col = lane%16 (+ nt*16)
  int crow = mtile * 16 + ((lane >> 4) ? 8 : 0);
  int ccol = lane & 15;
#pragma unroll
  for (int nt = 0; nt < NT_H; ++nt) {
    int col = nt * 16 + ccol;
    float bias = b1[col];
#pragma unroll
    for (int r = 0; r < 8; ++r)
      z[(size_t)(crow + r) * HID + col] = acc[nt][r] + bias;
  }
}

// ---------------- BatchNorm statistics ----------------
__global__ __launch_bounds__(256) void k_bn_reduce(const float* __restrict__ z,
                                                   float* __restrict__ bnsum,
                                                   float* __restrict__ bnsq) {
  const int ROWS = 196;  // 256 blocks * 196 >= 50000
  int col  = threadIdx.x & 63;
  int rgrp = threadIdx.x >> 6;  // 0..3
  int r0 = blockIdx.x * ROWS;
  int rend = min(r0 + ROWS, N_SPOTS);
  float s = 0.f, q = 0.f;
  for (int r = r0 + rgrp; r < rend; r += 4) {
    float v = z[(size_t)r * HID + col];
    s += v;
    q += v * v;
  }
  atomicAdd(bnsum + col, s);
  atomicAdd(bnsq + col, q);
}

__global__ void k_bn_final(const float* __restrict__ bnsum, const float* __restrict__ bnsq,
                           const float* __restrict__ gamma, const float* __restrict__ beta,
                           float* __restrict__ scale, float* __restrict__ shift) {
  int c = threadIdx.x;
  if (c >= HID) return;
  float inv_n = 1.f / (float)N_SPOTS;
  float mean = bnsum[c] * inv_n;
  float var  = bnsq[c] * inv_n - mean * mean;
  float sc = gamma[c] * rsqrtf(var + BN_EPS);
  scale[c] = sc;
  shift[c] = beta[c] - mean * sc;
}

// ---------------- h = relu(bn(z)) @ W2 + b2 ----------------
__global__ __launch_bounds__(256) void k_hidden(const float* __restrict__ z,
                                                const unsigned* __restrict__ wf,
                                                const float* __restrict__ b2,
                                                const float* __restrict__ scale,
                                                const float* __restrict__ shift,
                                                float* __restrict__ hout,
                                                unsigned short* __restrict__ hb) {
  int wave = threadIdx.x >> 5, lane = threadIdx.x & 31;
  int mtile = blockIdx.x * 8 + wave;
  if (mtile >= MTILES) return;
  int arow = mtile * 16 + (lane & 15);
  const float* zr = z + (size_t)arow * HID;
  int khalf = (lane & 16) ? 8 : 0;

  floatx8 acc[NT_H];
#pragma unroll
  for (int i = 0; i < NT_H; ++i) acc[i] = zero8();

#pragma unroll
  for (int t = 0; t < KT_H; ++t) {
    FragU bfr[NT_H];
#pragma unroll
    for (int nt = 0; nt < NT_H; ++nt) load_bfrag(wf, nt, t, KT_H, lane, bfr[nt]);

    FragU a;
#pragma unroll
    for (int h = 0; h < 2; ++h) {
      int c = t * 32 + khalf + h * 16;
#pragma unroll
      for (int j = 0; j < 4; ++j) {
        int k = c + 2 * j;
        float v0 = fmaxf(zr[k] * scale[k] + shift[k], 0.f);
        float v1 = fmaxf(zr[k + 1] * scale[k + 1] + shift[k + 1], 0.f);
        a.u[h * 4 + j] = packbf(v0, v1);
      }
    }
#pragma unroll
    for (int nt = 0; nt < NT_H; ++nt)
      acc[nt] = wmma_bf16(a, bfr[nt], acc[nt]);
  }

  int crow = mtile * 16 + ((lane >> 4) ? 8 : 0);
  int ccol = lane & 15;
#pragma unroll
  for (int nt = 0; nt < NT_H; ++nt) {
    int col = nt * 16 + ccol;
    float bias = b2[col];
#pragma unroll
    for (int r = 0; r < 8; ++r) {
      float v = acc[nt][r] + bias;
      size_t idx = (size_t)(crow + r) * HID + col;
      hout[idx] = v;          // f32 output "h"
      hb[idx]   = f2bf(v);    // bf16 copy for downstream GEMMs
    }
  }
}

// ---------------- out_bf16 = relu(A @ W + b), A 64-wide ----------------
// Mode 1 (cnt==null): A from bf16 buffer Ab.
// Mode 2 (cnt!=null): A row = aggf[row,:] / max(cnt[row],1)  (neighbor mean)
__global__ __launch_bounds__(256) void k_ff64(const unsigned short* __restrict__ Ab,
                                              const float* __restrict__ aggf,
                                              const float* __restrict__ cnt,
                                              const unsigned* __restrict__ wf,
                                              const float* __restrict__ bias,
                                              unsigned short* __restrict__ outb) {
  int wave = threadIdx.x >> 5, lane = threadIdx.x & 31;
  int mtile = blockIdx.x * 8 + wave;
  if (mtile >= MTILES) return;
  int arow = mtile * 16 + (lane & 15);
  int khalf = (lane & 16) ? 8 : 0;

  FragU a[KT_H];
  if (cnt == nullptr) {
    const unsigned short* rp = Ab + (size_t)arow * HID;
#pragma unroll
    for (int t = 0; t < KT_H; ++t) load_afrag_bf16(rp, t, khalf, a[t]);
  } else {
    float s = 1.f / fmaxf(cnt[arow], 1.f);
    const float* rp = aggf + (size_t)arow * HID;
#pragma unroll
    for (int t = 0; t < KT_H; ++t)
#pragma unroll
      for (int h = 0; h < 2; ++h) {
        int c = t * 32 + khalf + h * 16;
#pragma unroll
        for (int j = 0; j < 4; ++j)
          a[t].u[h * 4 + j] = packbf(rp[c + 2 * j] * s, rp[c + 2 * j + 1] * s);
      }
  }

  floatx8 acc[NT_H];
#pragma unroll
  for (int i = 0; i < NT_H; ++i) acc[i] = zero8();
#pragma unroll
  for (int t = 0; t < KT_H; ++t) {
    FragU bfr[NT_H];
#pragma unroll
    for (int nt = 0; nt < NT_H; ++nt) load_bfrag(wf, nt, t, KT_H, lane, bfr[nt]);
#pragma unroll
    for (int nt = 0; nt < NT_H; ++nt)
      acc[nt] = wmma_bf16(a[t], bfr[nt], acc[nt]);
  }

  int crow = mtile * 16 + ((lane >> 4) ? 8 : 0);
  int ccol = lane & 15;
#pragma unroll
  for (int nt = 0; nt < NT_H; ++nt) {
    int col = nt * 16 + ccol;
    float bb = bias[col];
#pragma unroll
    for (int r = 0; r < 8; ++r)
      outb[(size_t)(crow + r) * HID + col] = f2bf(fmaxf(acc[nt][r] + bb, 0.f));
  }
}

// ---------------- pos = relu(h@Wp1+bp1)@Wp2 + bp2 (WMMA->LDS->WMMA) ----------------
__global__ __launch_bounds__(256) void k_pos(const unsigned short* __restrict__ hb,
                                             const unsigned* __restrict__ wf1,
                                             const float* __restrict__ bp1,
                                             const unsigned* __restrict__ wf2,
                                             const float* __restrict__ bp2,
                                             float* __restrict__ pos) {
  __shared__ __align__(16) unsigned short sT[8][16 * HID];  // 16KB
  int wave = threadIdx.x >> 5, lane = threadIdx.x & 31;
  int mt = blockIdx.x * 8 + wave;
  bool valid = (mt < MTILES);
  int mtile = valid ? mt : MTILES - 1;  // clamp: all waves reach __syncthreads
  int khalf = (lane & 16) ? 8 : 0;
  int arow = mtile * 16 + (lane & 15);

  // layer 1
  FragU a[KT_H];
  const unsigned short* rp = hb + (size_t)arow * HID;
#pragma unroll
  for (int t = 0; t < KT_H; ++t) load_afrag_bf16(rp, t, khalf, a[t]);

  floatx8 acc[NT_H];
#pragma unroll
  for (int i = 0; i < NT_H; ++i) acc[i] = zero8();
#pragma unroll
  for (int t = 0; t < KT_H; ++t) {
    FragU bfr[NT_H];
#pragma unroll
    for (int nt = 0; nt < NT_H; ++nt) load_bfrag(wf1, nt, t, KT_H, lane, bfr[nt]);
#pragma unroll
    for (int nt = 0; nt < NT_H; ++nt)
      acc[nt] = wmma_bf16(a[t], bfr[nt], acc[nt]);
  }

  // spill t = relu(.) to LDS (row-major bf16 16x64) for re-fragmenting as A
  int lrow = (lane >> 4) ? 8 : 0;
  int lcol = lane & 15;
#pragma unroll
  for (int nt = 0; nt < NT_H; ++nt) {
    int col = nt * 16 + lcol;
    float bb = bp1[col];
#pragma unroll
    for (int r = 0; r < 8; ++r)
      sT[wave][(lrow + r) * HID + col] = f2bf(fmaxf(acc[nt][r] + bb, 0.f));
  }
  __syncthreads();

  // layer 2: 16x16 output, only columns 0..1 live (Wp2 fragment is zero-padded)
  floatx8 o = zero8();
#pragma unroll
  for (int t = 0; t < KT_H; ++t) {
    FragU a2, b;
    int c0 = t * 32 + khalf;
    const unsigned short* lp = &sT[wave][(lane & 15) * HID];
    a2.q[0] = *(const uint4*)(lp + c0);
    a2.q[1] = *(const uint4*)(lp + c0 + 16);
    load_bfrag(wf2, 0, t, KT_H, lane, b);
    o = wmma_bf16(a2, b, o);
  }
  int col = lane & 15;
  if (valid && col < 2) {
    float bb = bp2[col];
    int rb = mtile * 16 + ((lane >> 4) ? 8 : 0);
#pragma unroll
    for (int r = 0; r < 8; ++r)
      pos[(size_t)(rb + r) * 2 + col] = o[r] + bb;
  }
}

// ---------------- neighbor scatter: agg[dst] += h[src]; cnt[dst]++ ----------------
__global__ __launch_bounds__(256) void k_scatter(const int* __restrict__ src,
                                                 const int* __restrict__ dst,
                                                 const float* __restrict__ h,
                                                 float* __restrict__ agg,
                                                 float* __restrict__ cnt, int E) {
  long gid = (long)blockIdx.x * blockDim.x + threadIdx.x;
  int e = (int)(gid >> 4);
  int p = (int)(gid & 15);
  if (e >= E) return;
  int s = src[e], d = dst[e];
  float4 v = *(const float4*)(h + (size_t)s * HID + p * 4);
  float* ap = agg + (size_t)d * HID + p * 4;
  atomicAdd(ap + 0, v.x);
  atomicAdd(ap + 1, v.y);
  atomicAdd(ap + 2, v.z);
  atomicAdd(ap + 3, v.w);
  if (p == 0) atomicAdd(cnt + d, 1.f);
}

// ---------------- big output GEMM: out[N,2000] = A[N,64] @ W + b ----------------
__global__ __launch_bounds__(256) void k_biggemm(const unsigned short* __restrict__ Ab,
                                                 const unsigned* __restrict__ wf,
                                                 const float* __restrict__ bias,
                                                 float* __restrict__ out) {
  int wave = threadIdx.x >> 5, lane = threadIdx.x & 31;
  int mtile = blockIdx.x * 8 + wave;
  if (mtile >= MTILES) return;
  int nt0 = blockIdx.y * 25;  // 5 n-groups of 25 tiles = 125
  int khalf = (lane & 16) ? 8 : 0;
  int arow = mtile * 16 + (lane & 15);

  FragU a[KT_H];
  const unsigned short* rp = Ab + (size_t)arow * HID;
#pragma unroll
  for (int t = 0; t < KT_H; ++t) load_afrag_bf16(rp, t, khalf, a[t]);

  int crow = mtile * 16 + ((lane >> 4) ? 8 : 0);
  int ccol = lane & 15;

  // double-buffer B fragments across the n-tile loop
  FragU bq[2][KT_H];
#pragma unroll
  for (int t = 0; t < KT_H; ++t) load_bfrag(wf, nt0, t, KT_H, lane, bq[0][t]);

  for (int nt = nt0; nt < nt0 + 25; ++nt) {
    int cur = (nt - nt0) & 1;
    if (nt + 1 < nt0 + 25) {
#pragma unroll
      for (int t = 0; t < KT_H; ++t)
        load_bfrag(wf, nt + 1, t, KT_H, lane, bq[cur ^ 1][t]);
    }
    floatx8 acc = zero8();
#pragma unroll
    for (int t = 0; t < KT_H; ++t)
      acc = wmma_bf16(a[t], bq[cur][t], acc);
    int col = nt * 16 + ccol;
    float bb = bias[col];
#pragma unroll
    for (int r = 0; r < 8; ++r)
      out[(size_t)(crow + r) * N_GENES + col] = acc[r] + bb;
  }
}

// ---------------- host side ----------------
extern "C" void kernel_launch(void* const* d_in, const int* in_sizes, int n_in,
                              void* d_out, int out_size, void* d_ws, size_t ws_size,
                              hipStream_t stream) {
  const float* x     = (const float*)d_in[0];
  const int*   ei    = (const int*)d_in[1];
  const float* W1    = (const float*)d_in[2];
  const float* b1    = (const float*)d_in[3];
  const float* gamma = (const float*)d_in[4];
  const float* beta  = (const float*)d_in[5];
  const float* W2    = (const float*)d_in[6];
  const float* b2    = (const float*)d_in[7];
  const float* Wp1   = (const float*)d_in[8];
  const float* bp1   = (const float*)d_in[9];
  const float* Wp2   = (const float*)d_in[10];
  const float* bp2   = (const float*)d_in[11];
  const float* Ws1   = (const float*)d_in[12];
  const float* bs1   = (const float*)d_in[13];
  const float* Ws2   = (const float*)d_in[14];
  const float* bs2   = (const float*)d_in[15];
  const float* We1   = (const float*)d_in[16];
  const float* be1   = (const float*)d_in[17];
  const float* We2   = (const float*)d_in[18];
  const float* be2   = (const float*)d_in[19];
  const int E = in_sizes[1] / 2;
  const int* src = ei;
  const int* dst = ei + E;

  // output layout (tuple concatenated): h, pos, x_self, x_neighbor
  float* out   = (float*)d_out;
  float* h_out = out;
  float* pos   = out + (size_t)N_SPOTS * HID;
  float* xself = pos + (size_t)N_SPOTS * 2;
  float* xnb   = xself + (size_t)N_SPOTS * N_GENES;

  // workspace carve-up (256B aligned)
  char* w = (char*)d_ws;
  auto carve = [&](size_t bytes) { char* p = w; w += (bytes + 255) & ~(size_t)255; return p; };
  float*          zbuf   = (float*)carve((size_t)N_SPOTS * HID * 4);
  unsigned short* hb     = (unsigned short*)carve((size_t)N_SPOTS * HID * 2);
  unsigned short* aself  = (unsigned short*)carve((size_t)N_SPOTS * HID * 2);
  unsigned short* anb    = (unsigned short*)carve((size_t)N_SPOTS * HID * 2);
  float*          agg    = (float*)carve((size_t)N_SPOTS * HID * 4);
  float*          cnt    = (float*)carve((size_t)N_SPOTS * 4);
  float*          bnstat = (float*)carve(2 * HID * 4);  // sum | sumsq
  float*          bnaff  = (float*)carve(2 * HID * 4);  // scale | shift
  unsigned* f_W1  = (unsigned*)carve((size_t)KT_X * NT_H * 256 * 4);
  unsigned* f_W2  = (unsigned*)carve((size_t)KT_H * NT_H * 256 * 4);
  unsigned* f_Wp1 = (unsigned*)carve((size_t)KT_H * NT_H * 256 * 4);
  unsigned* f_Wp2 = (unsigned*)carve((size_t)KT_H * 1 * 256 * 4);
  unsigned* f_Ws1 = (unsigned*)carve((size_t)KT_H * NT_H * 256 * 4);
  unsigned* f_Ws2 = (unsigned*)carve((size_t)KT_H * NT_G * 256 * 4);
  unsigned* f_We1 = (unsigned*)carve((size_t)KT_H * NT_H * 256 * 4);
  unsigned* f_We2 = (unsigned*)carve((size_t)KT_H * NT_G * 256 * 4);

  const int WB = 256;
  const int MBLK = (MTILES + 7) / 8;  // 391

  // 0) per-call zeroing of accumulators (graph-replay safe)
  k_zero_f32<<<1, 128, 0, stream>>>(bnstat, 2 * HID);
  k_zero_f32<<<12500, WB, 0, stream>>>(agg, N_SPOTS * HID);
  k_zero_f32<<<196, WB, 0, stream>>>(cnt, N_SPOTS);

  // 1) weight -> bf16 B-fragment swizzle
  k_build_frag<<<252, WB, 0, stream>>>(W1,  N_GENES, HID,     KT_X, NT_H, f_W1);
  k_build_frag<<<8,   WB, 0, stream>>>(W2,  HID,     HID,     KT_H, NT_H, f_W2);
  k_build_frag<<<8,   WB, 0, stream>>>(Wp1, HID,     HID,     KT_H, NT_H, f_Wp1);
  k_build_frag<<<2,   WB, 0, stream>>>(Wp2, HID,     2,       KT_H, 1,    f_Wp2);
  k_build_frag<<<8,   WB, 0, stream>>>(Ws1, HID,     HID,     KT_H, NT_H, f_Ws1);
  k_build_frag<<<250, WB, 0, stream>>>(Ws2, HID,     N_GENES, KT_H, NT_G, f_Ws2);
  k_build_frag<<<8,   WB, 0, stream>>>(We1, HID,     HID,     KT_H, NT_H, f_We1);
  k_build_frag<<<250, WB, 0, stream>>>(We2, HID,     N_GENES, KT_H, NT_G, f_We2);

  // 2) z = x @ W1 + b1
  k_gemm1<<<MBLK, WB, 0, stream>>>(x, f_W1, b1, zbuf);

  // 3) BatchNorm stats -> affine
  k_bn_reduce<<<256, WB, 0, stream>>>(zbuf, bnstat, bnstat + HID);
  k_bn_final<<<1, 64, 0, stream>>>(bnstat, bnstat + HID, gamma, beta, bnaff, bnaff + HID);

  // 4) h = relu(bn(z)) @ W2 + b2   (f32 to d_out, bf16 copy for reuse)
  k_hidden<<<MBLK, WB, 0, stream>>>(zbuf, f_W2, b2, bnaff, bnaff + HID, h_out, hb);

  // 5) pos head
  k_pos<<<MBLK, WB, 0, stream>>>(hb, f_Wp1, bp1, f_Wp2, bp2, pos);

  // 6) self head: a = relu(h@Ws1+bs1); x_self = a@Ws2+bs2
  k_ff64<<<MBLK, WB, 0, stream>>>(hb, nullptr, nullptr, f_Ws1, bs1, aself);
  k_biggemm<<<dim3(MBLK, 5), WB, 0, stream>>>(aself, f_Ws2, bs2, xself);

  // 7) neighbor head: mean aggregation then MLP
  k_scatter<<<(int)(((long)E * 16 + WB - 1) / WB), WB, 0, stream>>>(src, dst, h_out, agg, cnt, E);
  k_ff64<<<MBLK, WB, 0, stream>>>(nullptr, agg, cnt, f_We1, be1, anb);
  k_biggemm<<<dim3(MBLK, 5), WB, 0, stream>>>(anb, f_We2, be2, xnb);
}
